// GNNLayer_14817637171801
// MI455X (gfx1250) — compile-verified, matrix operand order checked by reading, not run on hardware
//
#include <hip/hip_runtime.h>
#include <hip/hip_bf16.h>

// ---------------------------------------------------------------------------
// GNN layer for MI455X (gfx1250, wave32):
//   LE  = segment_sum(val * feats[src]) over dst        (L2-resident SpMM)
//   out = (LE+feats) @ W1^T + (LE*feats) @ W2^T + b1+b2 (WMMA bf16->f32)
//
// Weights are pre-converted to bf16 and pre-swizzled into the WMMA
// B-fragment lane layout once per call, so the hot dense loop is pure
// {global_load_b128 -> v_wmma_f32_16x16x32_bf16}.
// ---------------------------------------------------------------------------

#define FDIM 128

typedef __attribute__((ext_vector_type(16))) __bf16 v16bf;
typedef __attribute__((ext_vector_type(8)))  float  v8f;

// ---------------- Phase 0: zero the LE accumulator in workspace ------------
__global__ __launch_bounds__(256) void gnn_zero(float4* __restrict__ p, int n4) {
    int i = blockIdx.x * 256 + threadIdx.x;
    if (i < n4) p[i] = float4{0.f, 0.f, 0.f, 0.f};
}

// ---------------- Phase 0b: pack W1/W2 into WMMA B-fragment layout ---------
// Packed layout: v16bf unit index = ((mat*4 + ks)*8 + nt)*32 + lane.
// B 32x16 lane mapping (mirror of the 16-bit A 16x32 layout, ISA 7.12.2):
//   lane = n + 16h; element i covers K = (i<8) ? 8h+i : 16 + 8h + (i-8),
// and for B of W^T the (k, n) element is W[n][k] -> contiguous K runs of
// row-major W. 2048 fragment-lanes total, each writes one 32-byte v16bf.
__global__ __launch_bounds__(256) void gnn_packW(const float* __restrict__ W1,
                                                 const float* __restrict__ W2,
                                                 v16bf* __restrict__ packed) {
    const int id = blockIdx.x * 256 + threadIdx.x;
    if (id >= 2 * 4 * 8 * 32) return;
    const int lane = id & 31;
    const int nt   = (id >> 5) & 7;
    const int ks   = (id >> 8) & 3;
    const int mat  = id >> 10;
    const float* __restrict__ W = mat ? W2 : W1;
    const int n = lane & 15;
    const int h = lane >> 4;
    const float* __restrict__ wrow = W + (size_t)(nt * 16 + n) * FDIM + ks * 32;
    v16bf frag;
#pragma unroll
    for (int i = 0; i < 8; ++i) {
        frag[i]     = (__bf16)wrow[8 * h + i];
        frag[8 + i] = (__bf16)wrow[16 + 8 * h + i];
    }
    packed[id] = frag;
}

// ---------------- Phase 1: edge-parallel SpMM with f32 L2 atomics ----------
// One wave32 per edge; each lane owns a float4 slice of the 128-float row.
// feats (51MB) + LE (51MB) both fit in the 192MB L2, so the random gather /
// atomic scatter stays on-chip after first touch.
__global__ __launch_bounds__(256) void gnn_spmm(const int* __restrict__ esrc,
                                                const int* __restrict__ edst,
                                                const float* __restrict__ evals,
                                                const float* __restrict__ feats,
                                                float* __restrict__ LE,
                                                int nEdges) {
    const int lane = threadIdx.x & 31;
    const int e    = blockIdx.x * 8 + (threadIdx.x >> 5);
    if (e >= nEdges) return;
    const int   s = esrc[e];
    const int   d = edst[e];
    const float v = evals[e];
    const float4 f = *(const float4*)(feats + (size_t)s * FDIM + lane * 4);
    float* o = LE + (size_t)d * FDIM + lane * 4;
    atomicAdd(o + 0, v * f.x);
    atomicAdd(o + 1, v * f.y);
    atomicAdd(o + 2, v * f.z);
    atomicAdd(o + 3, v * f.w);
}

// ---------------- Phase 2: fused dual-GEMM via v_wmma_f32_16x16x32_bf16 ----
// One wave per 16-node tile (100000 = 6250 x 16). A-fragments for both
// GEMMs over the full K=128 are built once into registers (64 VGPRs); the
// pre-swizzled bf16 B-fragments stream straight from L2 with no conversion.
// 64 WMMAs per wave accumulate into one 16x128 f32 C tile.
// C/D f32 16x16 layout: VGPR r holds row r + 8h, column = lane & 15.
__global__ __launch_bounds__(256) void gnn_dense(const float* __restrict__ feats,
                                                 const float* __restrict__ LE,
                                                 const v16bf* __restrict__ packedW,
                                                 const float* __restrict__ b1,
                                                 const float* __restrict__ b2,
                                                 float* __restrict__ out,
                                                 int nTiles, int nNodes) {
    const int lane = threadIdx.x & 31;
    const int wave = threadIdx.x >> 5;
    const int tile = blockIdx.x * 8 + wave;
    if (tile >= nTiles) return;               // wave-uniform: EXEC stays all-1s

    const int m = lane & 15;                  // A row / B,C column within tile
    const int h = lane >> 4;                  // half select

    int node = tile * 16 + m;
    if (node > nNodes - 1) node = nNodes - 1; // clamp loads (stores guarded)
    const float* __restrict__ frow = feats + (size_t)node * FDIM;
    const float* __restrict__ lrow = LE    + (size_t)node * FDIM;

    // Build all A-fragments: a1 = bf16(LE + feats), a2 = bf16(LE * feats).
    // A 16x32 (16-bit) layout: lane = m + 16h; element i covers
    // K = (i<8) ? 8h+i : 16 + 8h + (i-8)  -> two contiguous 8-wide runs.
    v16bf a1[4], a2[4];
#pragma unroll
    for (int ks = 0; ks < 4; ++ks) {
        const int kb = ks * 32;
#pragma unroll
        for (int i = 0; i < 8; ++i) {
            float f0 = frow[kb + 8 * h + i];
            float l0 = lrow[kb + 8 * h + i];
            float f1 = frow[kb + 16 + 8 * h + i];
            float l1 = lrow[kb + 16 + 8 * h + i];
            a1[ks][i]     = (__bf16)(l0 + f0);
            a2[ks][i]     = (__bf16)(l0 * f0);
            a1[ks][8 + i] = (__bf16)(l1 + f1);
            a2[ks][8 + i] = (__bf16)(l1 * f1);
        }
    }

    v8f acc[8] = {};                          // 8 N-tiles of 16x16 f32

#pragma unroll
    for (int nt = 0; nt < 8; ++nt) {
#pragma unroll
        for (int ks = 0; ks < 4; ++ks) {
            // Pre-swizzled B fragments: one 32-byte vector load each.
            const v16bf bB1 = packedW[((0 * 4 + ks) * 8 + nt) * 32 + lane];
            const v16bf bB2 = packedW[((1 * 4 + ks) * 8 + nt) * 32 + lane];
            acc[nt] = __builtin_amdgcn_wmma_f32_16x16x32_bf16(
                false, a1[ks], false, bB1, (short)0, acc[nt], false, false);
            acc[nt] = __builtin_amdgcn_wmma_f32_16x16x32_bf16(
                false, a2[ks], false, bB2, (short)0, acc[nt], false, false);
        }
    }

    // Store C with fused bias (b1 + b2).
#pragma unroll
    for (int nt = 0; nt < 8; ++nt) {
        const int col  = nt * 16 + m;
        const float bs = b1[col] + b2[col];
#pragma unroll
        for (int r = 0; r < 8; ++r) {
            const int row = tile * 16 + r + 8 * h;
            if (row < nNodes)
                out[(size_t)row * FDIM + col] = acc[nt][r] + bs;
        }
    }
}

// ---------------------------------------------------------------------------
extern "C" void kernel_launch(void* const* d_in, const int* in_sizes, int n_in,
                              void* d_out, int out_size, void* d_ws, size_t ws_size,
                              hipStream_t stream) {
    const int*   edge_src  = (const int*)d_in[0];
    const int*   edge_dst  = (const int*)d_in[1];
    const float* edge_vals = (const float*)d_in[2];
    const float* feats     = (const float*)d_in[3];
    const float* W1        = (const float*)d_in[4];
    const float* b1        = (const float*)d_in[5];
    const float* W2        = (const float*)d_in[6];
    const float* b2        = (const float*)d_in[7];
    float*       out       = (float*)d_out;

    const int E = in_sizes[0];
    const int N = in_sizes[3] / FDIM;

    // Workspace layout: [ LE : N*FDIM f32 ][ packed bf16 weights : 64KB ]
    float* LE = (float*)d_ws;
    v16bf* packedW = (v16bf*)((char*)d_ws + (size_t)N * FDIM * sizeof(float));

    const int n4 = (N * FDIM) / 4;
    gnn_zero<<<(n4 + 255) / 256, 256, 0, stream>>>((float4*)LE, n4);

    gnn_packW<<<8, 256, 0, stream>>>(W1, W2, packedW);

    gnn_spmm<<<(E + 7) / 8, 256, 0, stream>>>(edge_src, edge_dst, edge_vals,
                                              feats, LE, E);

    const int nTiles = (N + 15) / 16;         // 6250 for N=100000 (exact)
    gnn_dense<<<(nTiles + 7) / 8, 256, 0, stream>>>(feats, LE, packedW, b1, b2,
                                                    out, nTiles, N);
}
